// MHDM_970662609221
// MI455X (gfx1250) — matrix operand order repaired and checked by reading, not compiled, not run on hardware
//
#include <hip/hip_runtime.h>

typedef _Float16 v16h __attribute__((ext_vector_type(16)));
typedef _Float16 v8h  __attribute__((ext_vector_type(8)));
typedef float    v8f  __attribute__((ext_vector_type(8)));

static constexpr int kB  = 2;
static constexpr int kN  = 2048;
static constexpr int kD  = 1024;
static constexpr int kH  = 16;
static constexpr int kDh = 64;          // head dim
static constexpr int kM  = kB * kN;     // 4096 token rows

__device__ __forceinline__ v8f zero8f() {
  v8f z;
  #pragma unroll
  for (int i = 0; i < 8; ++i) z[i] = 0.0f;
  return z;
}

// A-fragment (16x32 f16, M x K), contiguous K order:
//   lanes 0-15 (hf=0): K = {0..15} in elems 0..15
//   lanes 16-31 (hf=1): K = {16..31}
// 'row' points at this lane's A row at the desired K base (global or LDS).
__device__ __forceinline__ v16h wmma_a_frag(const _Float16* row, int hf) {
  v8h lo = *(const v8h*)(row + hf * 16);
  v8h hi = *(const v8h*)(row + hf * 16 + 8);
  v16h a;
  #pragma unroll
  for (int i = 0; i < 8; ++i) { a[i] = lo[i]; a[8 + i] = hi[i]; }
  return a;
}

__device__ __forceinline__ v8f wmma_f16(v16h a, v16h b, v8f c) {
  // (neg_a, A, neg_b, B, c_mod, C, reuse_a, reuse_b)
  return __builtin_amdgcn_wmma_f32_16x16x32_f16(false, a, false, b, (short)0, c,
                                                false, false);
}

// ---------------------------------------------------------------------------
__global__ void f32_to_f16_kernel(const float* __restrict__ in,
                                  _Float16* __restrict__ out, int n) {
  for (int i = blockIdx.x * blockDim.x + threadIdx.x; i < n;
       i += gridDim.x * blockDim.x)
    out[i] = (_Float16)in[i];
}

// ---------------------------------------------------------------------------
// C = A (kM x kD, f16 row-major) x Bt^T (Bt: kD x kD, row e holds W[e,:]).
// One wave computes a 32x64 output slab: 2 row sub-tiles x 4 col sub-tiles,
// 8 WMMAs per k-step for 6 b128-load-pairs (B fragments shared by both rows).
// mode 0: f16 out, (B,H,N,dh) layout  -> qk
// mode 1: f16 out, (B,H,dh,N) layout  -> v transposed (for PV B-fragments)
// mode 2: f32 out, (B,N,D) layout     -> final result
__global__ void __launch_bounds__(32)
proj_gemm_kernel(const _Float16* __restrict__ A, const _Float16* __restrict__ Bt,
                 void* __restrict__ out, int mode) {
  const int lane = threadIdx.x & 31;
  const int m  = lane & 15;
  const int hf = lane >> 4;
  const int row0 = blockIdx.y << 5;
  const int col0 = blockIdx.x << 6;

  v8f acc[2][4];
  #pragma unroll
  for (int rg = 0; rg < 2; ++rg)
    #pragma unroll
    for (int e = 0; e < 4; ++e) acc[rg][e] = zero8f();

  const _Float16* arow0 = A + (size_t)(row0 + m) * kD;
  const _Float16* arow1 = arow0 + (size_t)16 * kD;
  for (int k = 0; k < kD; k += 32) {
    v16h a0 = wmma_a_frag(arow0 + k, hf);
    v16h a1 = wmma_a_frag(arow1 + k, hf);
    #pragma unroll
    for (int e = 0; e < 4; ++e) {
      // B-frag: lane holds column (e*16+m) of B = row of Bt, K-contiguous.
      v16h b = *(const v16h*)(Bt + (size_t)(col0 + e * 16 + m) * kD + k + hf * 16);
      acc[0][e] = wmma_f16(a0, b, acc[0][e]);
      acc[1][e] = wmma_f16(a1, b, acc[1][e]);
    }
  }

  #pragma unroll
  for (int rg = 0; rg < 2; ++rg) {
    if (mode == 2) {
      float* of = (float*)out;
      #pragma unroll
      for (int e = 0; e < 4; ++e) {
        int col = col0 + e * 16 + m;
        #pragma unroll
        for (int r = 0; r < 8; ++r) {
          int grow = row0 + rg * 16 + r + (hf << 3);
          of[(size_t)grow * kD + col] = acc[rg][e][r];
        }
      }
    } else {
      _Float16* oh = (_Float16*)out;
      #pragma unroll
      for (int e = 0; e < 4; ++e) {
        int col = col0 + e * 16 + m;
        int hh = col >> 6;      // head
        int dd = col & 63;      // within-head dim
        #pragma unroll
        for (int r = 0; r < 8; ++r) {
          int grow = row0 + rg * 16 + r + (hf << 3);
          int gb = grow >> 11;          // / kN
          int n  = grow & (kN - 1);
          size_t idx = (mode == 0)
              ? (((size_t)(gb * kH + hh) * kN + n) * kDh + dd)    // (B,H,N,dh)
              : (((size_t)(gb * kH + hh) * kDh + dd) * kN + n);   // (B,H,dh,N)
          oh[idx] = (_Float16)acc[rg][e][r];
        }
      }
    }
  }
}

// ---------------------------------------------------------------------------
__global__ void q2_kernel(const _Float16* __restrict__ qk,
                          float* __restrict__ q2) {
  int i = blockIdx.x * blockDim.x + threadIdx.x;
  if (i >= kB * kH * kN) return;
  const _Float16* p = qk + (size_t)i * kDh;
  float s = 0.f;
  #pragma unroll 8
  for (int d = 0; d < kDh; ++d) { float v = (float)p[d]; s += v * v; }
  q2[i] = s;
}

// ---------------------------------------------------------------------------
// Flash-style distance attention. Block = 8 waves on one (b,h); each wave owns
// 16 query rows; the block cooperatively double-buffers each 32-key K / V^T
// chunk in LDS (8x reuse vs. per-wave global streaming).
// Computes S^T = K x Q^T so that (a) each lane's values share one query ->
// softmax reductions are a single shfl_xor(16), and (b) the f32 C-layout of
// the two 16-key tiles packs directly into the f16 A-fragment for P x V.
__global__ void __launch_bounds__(256)
dist_attn_kernel(const _Float16* __restrict__ qk, const _Float16* __restrict__ vT,
                 const float* __restrict__ q2, _Float16* __restrict__ attnO) {
  __shared__ alignas(64) _Float16 kbuf[2][32 * kDh];   // [key][d], chunk-contig
  __shared__ alignas(64) _Float16 vbuf[2][kDh * 32];   // [d][key]

  const int tid  = threadIdx.x;
  const int wave = tid >> 5;
  const int lane = tid & 31;
  const int m  = lane & 15;
  const int hf = lane >> 4;

  constexpr int nblk = kN / 128;            // query blocks per head = 16
  const int tb = blockIdx.x % nblk;
  const int bh = blockIdx.x / nblk;
  const int gb = bh >> 4;                   // / kH
  const int h  = bh & (kH - 1);
  const _Float16* qkH = qk + (size_t)bh * kN * kDh;
  const _Float16* vTH = vT + (size_t)bh * kDh * kN;
  const float* q2H = q2 + (size_t)bh * kN;
  const int q0 = tb * 128 + wave * 16;

  // Q as B-fragments (column = query m, K = head dim), loaded once per wave.
  v16h qb[2];
  #pragma unroll
  for (int c = 0; c < 2; ++c)
    qb[c] = *(const v16h*)(qkH + (size_t)(q0 + m) * kDh + c * 32 + hf * 16);

  // Cooperative staging: one b128 per thread per buffer.
  const int koff = tid * 8;                 // K chunk is contiguous (32*64 h)
  const int vd   = tid >> 2;                // 0..63 (d row of vT)
  const int vseg = (tid & 3) * 8;           // 4 x 16B segments per row

  v8h kr = *(const v8h*)(qkH + koff);                       // chunk j=0
  v8h vr = *(const v8h*)(vTH + (size_t)vd * kN + vseg);
  *(v8h*)(&kbuf[0][koff]) = kr;
  *(v8h*)(&vbuf[0][vd * 32 + vseg]) = vr;
  __syncthreads();

  v8f o[4];
  #pragma unroll
  for (int n = 0; n < 4; ++n) o[n] = zero8f();
  float m_run = -1e30f, l_run = 0.f;

  for (int j = 0; j < kN; j += 32) {
    const int cur = (j >> 5) & 1;
    const int nxt = cur ^ 1;
    const bool more = (j + 32) < kN;
    if (more) {   // prefetch next chunk into registers (overlaps compute)
      kr = *(const v8h*)(qkH + (size_t)(j + 32) * kDh + koff);
      vr = *(const v8h*)(vTH + (size_t)vd * kN + (j + 32) + vseg);
    }

    // S^T tiles from LDS: keys j..j+15 (t=0) / j+16..j+31 (t=1), queries cols.
    v8f st[2];
    #pragma unroll
    for (int t = 0; t < 2; ++t) {
      st[t] = zero8f();
      const _Float16* krow = &kbuf[cur][(t * 16 + m) * kDh];
      #pragma unroll
      for (int c = 0; c < 2; ++c)
        st[t] = wmma_f16(wmma_a_frag(krow + c * 32, hf), qb[c], st[t]);
    }
    // logits^T = 2*dot - q2[key] (row-constant -q2[query] dropped: softmax-inv.)
    v8f q2v0 = *(const v8f*)(q2H + j + hf * 8);
    v8f q2v1 = *(const v8f*)(q2H + j + 16 + hf * 8);
    v8f t0, t1;
    float lmax = -1e30f;
    #pragma unroll
    for (int r = 0; r < 8; ++r) {
      t0[r] = 2.f * st[0][r] - q2v0[r];
      t1[r] = 2.f * st[1][r] - q2v1[r];
      lmax = fmaxf(lmax, fmaxf(t0[r], t1[r]));
    }
    lmax = fmaxf(lmax, __shfl_xor(lmax, 16, 32));   // combine key halves
    float mnew  = fmaxf(m_run, lmax);
    float alpha = __expf(m_run - mnew);
    float lsum = 0.f;
    v16h pa;                                        // P as A-fragment, directly
    #pragma unroll
    for (int r = 0; r < 8; ++r) {
      float p0 = __expf(t0[r] - mnew);
      float p1 = __expf(t1[r] - mnew);
      lsum += p0 + p1;
      pa[r]     = (_Float16)p0;                     // keys {0..7}/{8..15}
      pa[8 + r] = (_Float16)p1;                     // keys {16..23}/{24..31}
    }
    lsum += __shfl_xor(lsum, 16, 32);
    l_run = l_run * alpha + lsum;
    m_run = mnew;
    // Redistribute per-query alpha to C-layout rows (query = r + 8*hf).
    float af[8];
    #pragma unroll
    for (int r = 0; r < 8; ++r) af[r] = __shfl(alpha, r + (hf << 3), 32);
    #pragma unroll
    for (int n = 0; n < 4; ++n) {
      #pragma unroll
      for (int r = 0; r < 8; ++r) o[n][r] *= af[r];
      // V B-frag from LDS: lane holds d-column (n*16+m), keys {8hf..+7},{16+8hf..+7}.
      v8h vlo = *(const v8h*)(&vbuf[cur][(n * 16 + m) * 32 + hf * 8]);
      v8h vhi = *(const v8h*)(&vbuf[cur][(n * 16 + m) * 32 + 16 + hf * 8]);
      v16h vb = __builtin_shufflevector(vlo, vhi, 0, 1, 2, 3, 4, 5, 6, 7, 8, 9, 10, 11, 12, 13, 14, 15);
      o[n] = wmma_f16(pa, vb, o[n]);
    }

    if (more) {   // publish next chunk, then one barrier per iteration
      *(v8h*)(&kbuf[nxt][koff]) = kr;
      *(v8h*)(&vbuf[nxt][vd * 32 + vseg]) = vr;
    }
    __syncthreads();
  }

  float lf[8];
  #pragma unroll
  for (int r = 0; r < 8; ++r) lf[r] = 1.f / __shfl(l_run, r + (hf << 3), 32);
  #pragma unroll
  for (int n = 0; n < 4; ++n) {
    int col = h * kDh + n * 16 + m;
    #pragma unroll
    for (int r = 0; r < 8; ++r) {
      int q = q0 + r + (hf << 3);
      attnO[(size_t)(gb * kN + q) * kD + col] = (_Float16)(o[n][r] * lf[r]);
    }
  }
}

// ---------------------------------------------------------------------------
extern "C" void kernel_launch(void* const* d_in, const int* in_sizes, int n_in,
                              void* d_out, int out_size, void* d_ws, size_t ws_size,
                              hipStream_t stream) {
  (void)in_sizes; (void)n_in; (void)out_size; (void)ws_size;
  const float* x    = (const float*)d_in[0];
  const float* Wqk  = (const float*)d_in[1];
  const float* Wv   = (const float*)d_in[2];
  const float* Wout = (const float*)d_in[3];

  char* ws = (char*)d_ws;
  size_t off = 0;
  auto take = [&](size_t bytes) -> void* {
    void* p = ws + off;
    off += (bytes + 255) & ~(size_t)255;
    return p;
  };
  _Float16* xh    = (_Float16*)take((size_t)kM * kD * 2);
  _Float16* wqkh  = (_Float16*)take((size_t)kD * kD * 2);
  _Float16* wvh   = (_Float16*)take((size_t)kD * kD * 2);
  _Float16* wouth = (_Float16*)take((size_t)kD * kD * 2);
  _Float16* qkh   = (_Float16*)take((size_t)kM * kD * 2);   // (B,H,N,dh)
  _Float16* vTh   = (_Float16*)take((size_t)kM * kD * 2);   // (B,H,dh,N)
  float*    q2b   = (float*)take((size_t)kB * kH * kN * 4);
  _Float16* aOh   = (_Float16*)take((size_t)kM * kD * 2);   // (B,N,D)

  f32_to_f16_kernel<<<2048, 256, 0, stream>>>(x,    xh,    kM * kD);
  f32_to_f16_kernel<<<1024, 256, 0, stream>>>(Wqk,  wqkh,  kD * kD);
  f32_to_f16_kernel<<<1024, 256, 0, stream>>>(Wv,   wvh,   kD * kD);
  f32_to_f16_kernel<<<1024, 256, 0, stream>>>(Wout, wouth, kD * kD);

  dim3 gg(kD / 64, kM / 32);                                 // (16, 128)
  proj_gemm_kernel<<<gg, 32, 0, stream>>>(xh, wqkh, qkh, 0);
  proj_gemm_kernel<<<gg, 32, 0, stream>>>(xh, wvh,  vTh, 1);
  q2_kernel<<<(kB * kH * kN + 255) / 256, 256, 0, stream>>>(qkh, q2b);
  dist_attn_kernel<<<kB * kH * (kN / 128), 256, 0, stream>>>(qkh, vTh, q2b, aOh);
  proj_gemm_kernel<<<gg, 32, 0, stream>>>(aOh, wouth, d_out, 2);
}
